// GRUModel_53678501265523
// MI455X (gfx1250) — matrix operand order repaired and checked by reading, not compiled, not run on hardware
//
#include <hip/hip_runtime.h>
#include <hip/hip_bf16.h>
#include <stdint.h>

// ---------------------------------------------------------------------------
// Types for CDNA5 WMMA (wave32, 16x16x32 bf16 -> f32)
// ---------------------------------------------------------------------------
typedef __bf16 bf16;
typedef __attribute__((ext_vector_type(16))) __bf16 v16bf;
typedef __attribute__((ext_vector_type(8)))  float  v8f;
typedef __attribute__((ext_vector_type(4)))  unsigned u32x4;

#define INPUT_DIM 300
#define INPUT_PAD 320
#define HIDDEN    512
#define GDIM      1536        // 3*HIDDEN gates
#define BATCH     128
#define TSTEPS    256
#define REC_BLOCKS 32         // 256 waves: 8 mt (16 rows) x 32 jblk (48 permuted cols)

union FragU { v16bf v; u32x4 q[2]; };

__device__ inline bf16 f2bf(float f) {
  unsigned u = __builtin_bit_cast(unsigned, f);
  unsigned r = u + 0x7fffu + ((u >> 16) & 1u);      // round-to-nearest-even
  unsigned short h = (unsigned short)(r >> 16);
  return __builtin_bit_cast(bf16, h);
}
__device__ inline float bf2f(bf16 b) {
  unsigned short h = __builtin_bit_cast(unsigned short, b);
  unsigned u = ((unsigned)h) << 16;
  return __builtin_bit_cast(float, u);
}
__device__ inline float sigmoidf_(float x) { return 1.0f / (1.0f + __expf(-x)); }

// A fragment: 16x32 bf16 tile, rows m0..m0+15, k0..k0+31, row-major src (global).
__device__ inline v16bf load_a_frag(const bf16* A, int lda, int m0, int k0, int lane) {
  const bf16* p = A + (size_t)(m0 + (lane & 15)) * lda + k0 + ((lane >> 4) << 3);
  FragU f;
  f.q[0] = *(const u32x4*)(p);
  f.q[1] = *(const u32x4*)(p + 16);
  return f.v;
}
// A fragment from LDS ([rows][512] row-major) -> ds_load_b128 x2
__device__ inline v16bf load_a_lds(const bf16* Al, int m0, int k0, int lane) {
  const bf16* p = Al + (size_t)(m0 + (lane & 15)) * HIDDEN + k0 + ((lane >> 4) << 3);
  FragU f;
  f.q[0] = *(const u32x4*)(p);
  f.q[1] = *(const u32x4*)(p + 16);
  return f.v;
}
// B fragment: 32x16 (KxN) from weights stored [N][K] row-major (global).
__device__ inline v16bf load_b_frag(const bf16* W, int ldw, int n0, int k0, int lane) {
  const bf16* p = W + (size_t)(n0 + (lane & 15)) * ldw + k0 + ((lane >> 4) << 3);
  FragU f;
  f.q[0] = *(const u32x4*)(p);
  f.q[1] = *(const u32x4*)(p + 16);
  return f.v;
}
// B fragment from LDS (same [N][512] row-major layout) -> ds_load_b128 x2
__device__ inline v16bf load_b_lds(const bf16* Wl, int n0, int k0, int lane) {
  const bf16* p = Wl + (size_t)(n0 + (lane & 15)) * HIDDEN + k0 + ((lane >> 4) << 3);
  FragU f;
  f.q[0] = *(const u32x4*)(p);
  f.q[1] = *(const u32x4*)(p + 16);
  return f.v;
}

__device__ inline v8f wmma_bf16(v16bf a, v16bf b, v8f c) {
  return __builtin_amdgcn_wmma_f32_16x16x32_bf16(false, a, false, b, (short)0, c, false, false);
}

// Async global->LDS 16B/lane copy (ASYNCcnt path)
__device__ inline void async_g2l_b128(unsigned lds_off, const void* gaddr) {
  asm volatile("global_load_async_to_lds_b128 %0, %1, off"
               :: "v"(lds_off), "v"(gaddr) : "memory");
}
__device__ inline void wait_async0() {
  asm volatile("s_wait_asynccnt 0x0" ::: "memory");
}

// Software grid barrier: one-shot counters, zeroed before each persistent launch.
__device__ inline void grid_barrier(unsigned* bar, unsigned nblk) {
  __syncthreads();
  if (threadIdx.x == 0) {
    __threadfence();
    atomicAdd(bar, 1u);
    volatile unsigned* vb = bar;
    while (*vb < nblk) { __builtin_amdgcn_s_sleep(2); }
    __threadfence();
  }
  __syncthreads();
}

// ---------------------------------------------------------------------------
// Converters
// ---------------------------------------------------------------------------
__global__ void cvt_pad(const float* __restrict__ in, bf16* __restrict__ out,
                        int R, int C, int Cp) {
  long total = (long)R * Cp;
  long stride = (long)gridDim.x * blockDim.x;
  for (long i = (long)blockIdx.x * blockDim.x + threadIdx.x; i < total; i += stride) {
    long r = i / Cp;
    int  c = (int)(i - r * Cp);
    float v = (c < C) ? in[r * C + c] : 0.0f;
    out[i] = f2bf(v);
  }
}

// gate-permuted weight convert: out row n <- in row g*512 + jblk*16 + jr,
// where jblk = n/48, g = (n%48)/16, jr = n%16.  (1536 rows)
__global__ void cvt_permute_gate(const float* __restrict__ in, bf16* __restrict__ out,
                                 int C, int Cp) {
  long total = (long)GDIM * Cp;
  long stride = (long)gridDim.x * blockDim.x;
  for (long i = (long)blockIdx.x * blockDim.x + threadIdx.x; i < total; i += stride) {
    int n = (int)(i / Cp);
    int c = (int)(i - (long)n * Cp);
    int jblk = n / 48, rem = n % 48;
    int g = rem >> 4, jr = rem & 15;
    int src = g * HIDDEN + jblk * 16 + jr;
    float v = (c < C) ? in[(size_t)src * C + c] : 0.0f;
    out[i] = f2bf(v);
  }
}

__global__ void permute_bias(const float* __restrict__ in, float* __restrict__ out) {
  int n = blockIdx.x * blockDim.x + threadIdx.x;
  if (n < GDIM) {
    int jblk = n / 48, rem = n % 48;
    int g = rem >> 4, jr = rem & 15;
    out[n] = in[g * HIDDEN + jblk * 16 + jr];
  }
}

__global__ void zero_u32(unsigned* __restrict__ p, int n) {
  for (int i = blockIdx.x * blockDim.x + threadIdx.x; i < n;
       i += gridDim.x * blockDim.x)
    p[i] = 0u;
}

// ---------------------------------------------------------------------------
// Projection GEMM: out[m][n] = sum_k A[m][k] * W[n][k] + bias[n]   (bf16 out)
// Block = 256 thr (8 waves): 64(M) x 256(N) tile; wave tile 32x64 = 2x4 WMMAs.
// W / bias are already gate-permuted, so `out` has permuted columns.
// ---------------------------------------------------------------------------
__global__ __launch_bounds__(256) void gemm_bias_bf16(
    const bf16* __restrict__ A, int lda,
    const bf16* __restrict__ W, int K,
    const float* __restrict__ bias,
    bf16* __restrict__ out, int ldo) {
  const int lane = threadIdx.x & 31;
  const int wave = threadIdx.x >> 5;
  const int m0 = blockIdx.y * 64 + (wave >> 2) * 32;
  const int n0 = blockIdx.x * 256 + (wave & 3) * 64;

  v8f c[2][4] = {};

  for (int k0 = 0; k0 < K; k0 += 32) {
    if (k0 + 32 < K)
      __builtin_prefetch(A + (size_t)(m0 + (lane & 15)) * lda + k0 + 32, 0, 1);
    v16bf a0 = load_a_frag(A, lda, m0,      k0, lane);
    v16bf a1 = load_a_frag(A, lda, m0 + 16, k0, lane);
    v16bf b0 = load_b_frag(W, K, n0,      k0, lane);
    v16bf b1 = load_b_frag(W, K, n0 + 16, k0, lane);
    v16bf b2 = load_b_frag(W, K, n0 + 32, k0, lane);
    v16bf b3 = load_b_frag(W, K, n0 + 48, k0, lane);
    c[0][0] = wmma_bf16(a0, b0, c[0][0]);
    c[0][1] = wmma_bf16(a0, b1, c[0][1]);
    c[0][2] = wmma_bf16(a0, b2, c[0][2]);
    c[0][3] = wmma_bf16(a0, b3, c[0][3]);
    c[1][0] = wmma_bf16(a1, b0, c[1][0]);
    c[1][1] = wmma_bf16(a1, b1, c[1][1]);
    c[1][2] = wmma_bf16(a1, b2, c[1][2]);
    c[1][3] = wmma_bf16(a1, b3, c[1][3]);
  }

  const int rbase = (lane >> 4) << 3;
  const int cn = lane & 15;
#pragma unroll
  for (int i = 0; i < 2; ++i)
#pragma unroll
    for (int j = 0; j < 4; ++j) {
      int n = n0 + j * 16 + cn;
      float bz = bias[n];
#pragma unroll
      for (int v = 0; v < 8; ++v) {
        int m = m0 + i * 16 + rbase + v;
        out[(size_t)m * ldo + n] = f2bf(c[i][j][v] + bz);
      }
    }
}

// ---------------------------------------------------------------------------
// Persistent GRU recurrence, gate-permuted weights, fused register gates.
//   - 32 blocks x 8 waves; wave tile = 16 rows x 48 permuted cols
//   - w_hh slab (192 rows x 512 = 192KB) staged into LDS ONCE (async DMA)
//   - per-step: the block's 32 hidden-state rows (32KB) staged into LDS (async
//     DMA burst), so the K-loop is pure ds_load -> v_wmma
//   - double-buffered bf16 hidden state -> ONE grid barrier per timestep
//   - LDS usage: 192KB (W) + 32KB (A) = 224KB of the 320KB WGP pool
// ---------------------------------------------------------------------------
__global__ __launch_bounds__(256) void gru_rec(
    const bf16* __restrict__ whp,   // [1536][512] bf16, gate-permuted rows
    const float* __restrict__ bhh,  // [1536] f32, original gate-major order
    const bf16* __restrict__ xg,    // [T*128][1536] bf16, permuted cols (b_ih included)
    bf16* __restrict__ h_a,         // [128][512] ping
    bf16* __restrict__ h_b,         // [128][512] pong
    float* __restrict__ h32,        // [128][512] f32 master state
    bf16* __restrict__ ys,          // [T*128][512] std layout, or nullptr
    unsigned* __restrict__ bars) {
  extern __shared__ bf16 lw[];                 // [192][512] weights + [32][512] A
  bf16* la = lw + (size_t)192 * HIDDEN;        // per-step A slab (32 rows)

  const int lane = threadIdx.x & 31;
  const int wave = threadIdx.x >> 5;
  const int mtg = blockIdx.x >> 3;          // 0..3  (block's pair of mt)
  const int jbg = blockIdx.x & 7;           // 0..7  (block's group of 4 jblk)
  const int jbl = wave & 3;                 // local jblk in LDS slab
  const int jb  = jbg * 4 + jbl;            // 0..31
  const int m0l = (wave >> 2) * 16;         // A row offset inside LDS slab
  const int m0g = mtg * 32 + m0l;           // global row base of this wave
  const int n0  = jb * 48;
  const int tid = blockIdx.x * 256 + threadIdx.x;
  const int nth = REC_BLOCKS * 256;
  int bs = 0;

  const unsigned lds0  = (unsigned)(size_t)(&lw[0]);
  const unsigned ldsA  = (unsigned)(size_t)(la);

  // ---- stage this block's 192 permuted weight rows into LDS (async DMA) ----
  {
    const char* gsrc = (const char*)(whp + (size_t)(jbg * 192) * HIDDEN);
#pragma unroll 4
    for (int k = 0; k < 48; ++k) {                  // 12288 x 16B chunks / block
      int chunk = threadIdx.x + 256 * k;
      async_g2l_b128(lds0 + chunk * 16, gsrc + (size_t)chunk * 16);
    }
  }

  // ---- h0 = 0 (overlaps with weight DMA) ----
  for (int i = tid; i < BATCH * HIDDEN; i += nth) {
    h32[i] = 0.0f;
    h_a[i] = f2bf(0.0f);
  }
  wait_async0();
  __syncthreads();
  grid_barrier(bars + bs++, REC_BLOCKS);

  const int rbase = (lane >> 4) << 3;
  const int cn = lane & 15;
  const int j = jb * 16 + cn;               // hidden unit owned by this lane/wave
  const bf16* lwf = lw + (size_t)(jbl * 48) * HIDDEN;
  const float bh_r = bhh[j];
  const float bh_z = bhh[HIDDEN + j];
  const float bh_n = bhh[2 * HIDDEN + j];

  for (int t = 0; t < TSTEPS; ++t) {
    const bf16* hin  = (t & 1) ? h_b : h_a;
    bf16*       hout = (t & 1) ? h_a : h_b;

    // ---- async-stage the block's 32 hidden rows for this step ----
    {
      const char* ga = (const char*)(hin + (size_t)(mtg * 32) * HIDDEN);
#pragma unroll
      for (int k = 0; k < 8; ++k) {                 // 2048 x 16B chunks / block
        int chunk = threadIdx.x + 256 * k;
        async_g2l_b128(ldsA + chunk * 16, ga + (size_t)chunk * 16);
      }
      wait_async0();
      __syncthreads();
    }

    // warm next step's xg slice in cache while we compute
    if (t + 1 < TSTEPS)
      __builtin_prefetch(xg + (size_t)((t + 1) * BATCH + m0g + rbase) * GDIM + n0, 0, 1);

    // ---- hidden GEMM: 16x48 tile, A and B from LDS ----
    v8f c0 = {}, c1 = {}, c2 = {};
    for (int k0 = 0; k0 < HIDDEN; k0 += 32) {
      v16bf a  = load_a_lds(la, m0l, k0, lane);
      v16bf b0 = load_b_lds(lwf, 0,  k0, lane);
      v16bf b1 = load_b_lds(lwf, 16, k0, lane);
      v16bf b2 = load_b_lds(lwf, 32, k0, lane);
      c0 = wmma_bf16(a, b0, c0);
      c1 = wmma_bf16(a, b1, c1);
      c2 = wmma_bf16(a, b2, c2);
    }

    // ---- gates fused in registers: c0=hr, c1=hz, c2=hn for unit j ----
#pragma unroll
    for (int v = 0; v < 8; ++v) {
      int m = m0g + rbase + v;
      const bf16* xm = xg + (size_t)(t * BATCH + m) * GDIM + n0;
      float xr = bf2f(xm[cn]);
      float xz = bf2f(xm[16 + cn]);
      float xn = bf2f(xm[32 + cn]);
      float r  = sigmoidf_(xr + c0[v] + bh_r);
      float z  = sigmoidf_(xz + c1[v] + bh_z);
      float nn = tanhf(xn + r * (c2[v] + bh_n));
      int hidx = m * HIDDEN + j;
      float hp = h32[hidx];
      float hnew = (1.0f - z) * nn + z * hp;
      h32[hidx] = hnew;
      hout[hidx] = f2bf(hnew);
      if (ys) ys[(size_t)(t * BATCH + m) * HIDDEN + j] = f2bf(hnew);
    }
    grid_barrier(bars + bs++, REC_BLOCKS);   // hout visible before next read
  }
}

// ---------------------------------------------------------------------------
// Head: BN -> fc1(512->256) -> ReLU -> LayerNorm(256) -> fc2(256->3)
// ---------------------------------------------------------------------------
__global__ __launch_bounds__(256) void head_kernel(
    const float* __restrict__ h32,
    const float* __restrict__ bn_w, const float* __restrict__ bn_b,
    const float* __restrict__ bn_mean, const float* __restrict__ bn_var,
    const float* __restrict__ fc1_w, const float* __restrict__ fc1_b,
    const float* __restrict__ ln_w, const float* __restrict__ ln_b,
    const float* __restrict__ fc2_w, const float* __restrict__ fc2_b,
    float* __restrict__ out) {
  __shared__ float sh[HIDDEN];
  __shared__ float red[256];
  __shared__ float o3[3];
  const int b = blockIdx.x;
  const int t = threadIdx.x;

  for (int k = t; k < HIDDEN; k += 256) {
    float x = h32[(size_t)b * HIDDEN + k];
    sh[k] = (x - bn_mean[k]) * rsqrtf(bn_var[k] + 1e-5f) * bn_w[k] + bn_b[k];
  }
  __syncthreads();

  float acc = fc1_b[t];
  const float* wr = fc1_w + (size_t)t * HIDDEN;
  for (int k = 0; k < HIDDEN; ++k) acc += sh[k] * wr[k];
  acc = fmaxf(acc, 0.0f);

  red[t] = acc;
  __syncthreads();
  for (int s = 128; s > 0; s >>= 1) {
    if (t < s) red[t] += red[t + s];
    __syncthreads();
  }
  float mu = red[0] * (1.0f / 256.0f);
  __syncthreads();
  float d = acc - mu;
  red[t] = d * d;
  __syncthreads();
  for (int s = 128; s > 0; s >>= 1) {
    if (t < s) red[t] += red[t + s];
    __syncthreads();
  }
  float var = red[0] * (1.0f / 256.0f);
  float vn = d * rsqrtf(var + 1e-5f) * ln_w[t] + ln_b[t];

  if (t < 3) o3[t] = fc2_b[t];
  __syncthreads();
#pragma unroll
  for (int o = 0; o < 3; ++o) atomicAdd(&o3[o], vn * fc2_w[o * 256 + t]);
  __syncthreads();
  if (t < 3) out[b * 3 + t] = o3[t];
}

// ---------------------------------------------------------------------------
// Launcher
// ---------------------------------------------------------------------------
extern "C" void kernel_launch(void* const* d_in, const int* in_sizes, int n_in,
                              void* d_out, int out_size, void* d_ws, size_t ws_size,
                              hipStream_t stream) {
  const float* x       = (const float*)d_in[0];
  const float* w_ih0   = (const float*)d_in[1];
  const float* w_hh0   = (const float*)d_in[2];
  const float* b_ih0   = (const float*)d_in[3];
  const float* b_hh0   = (const float*)d_in[4];
  const float* w_ih1   = (const float*)d_in[5];
  const float* w_hh1   = (const float*)d_in[6];
  const float* b_ih1   = (const float*)d_in[7];
  const float* b_hh1   = (const float*)d_in[8];
  const float* bn_w    = (const float*)d_in[9];
  const float* bn_b    = (const float*)d_in[10];
  const float* bn_mean = (const float*)d_in[11];
  const float* bn_var  = (const float*)d_in[12];
  const float* fc1_w   = (const float*)d_in[13];
  const float* fc1_b   = (const float*)d_in[14];
  const float* ln_w    = (const float*)d_in[15];
  const float* ln_b    = (const float*)d_in[16];
  const float* fc2_w   = (const float*)d_in[17];
  const float* fc2_b   = (const float*)d_in[18];
  float* out = (float*)d_out;
  (void)in_sizes; (void)n_in; (void)out_size; (void)ws_size;

  char* base = (char*)d_ws;
  size_t off = 0;
  auto take = [&](size_t bytes) -> void* {
    void* p = base + off;
    off += (bytes + 255) & ~(size_t)255;
    return p;
  };

  const int M = BATCH * TSTEPS;  // 32768 rows, t-major: m = t*B + b
  bf16* xbf   = (bf16*)take((size_t)M * INPUT_PAD * sizeof(bf16));
  bf16* wih0  = (bf16*)take((size_t)GDIM * INPUT_PAD * sizeof(bf16));
  bf16* whh0  = (bf16*)take((size_t)GDIM * HIDDEN * sizeof(bf16));
  bf16* wih1  = (bf16*)take((size_t)GDIM * HIDDEN * sizeof(bf16));
  bf16* whh1  = (bf16*)take((size_t)GDIM * HIDDEN * sizeof(bf16));
  bf16* xgb   = (bf16*)take((size_t)M * GDIM * sizeof(bf16));   // reused for both layers
  bf16* ys0   = (bf16*)take((size_t)M * HIDDEN * sizeof(bf16));
  bf16* h_a   = (bf16*)take((size_t)BATCH * HIDDEN * sizeof(bf16));
  bf16* h_b   = (bf16*)take((size_t)BATCH * HIDDEN * sizeof(bf16));
  float* h32  = (float*)take((size_t)BATCH * HIDDEN * sizeof(float));
  float* pb0  = (float*)take(GDIM * sizeof(float));   // permuted b_ih_l0
  float* pb1  = (float*)take(GDIM * sizeof(float));   // permuted b_ih_l1
  unsigned* bars = (unsigned*)take(600 * sizeof(unsigned));

  // Convert / permute weights and biases to the gate-interleaved layout
  cvt_pad<<<2048, 256, 0, stream>>>(x, xbf, M, INPUT_DIM, INPUT_PAD);
  cvt_permute_gate<<<256, 256, 0, stream>>>(w_ih0, wih0, INPUT_DIM, INPUT_PAD);
  cvt_permute_gate<<<256, 256, 0, stream>>>(w_hh0, whh0, HIDDEN, HIDDEN);
  cvt_permute_gate<<<256, 256, 0, stream>>>(w_ih1, wih1, HIDDEN, HIDDEN);
  cvt_permute_gate<<<256, 256, 0, stream>>>(w_hh1, whh1, HIDDEN, HIDDEN);
  permute_bias<<<6, 256, 0, stream>>>(b_ih0, pb0);
  permute_bias<<<6, 256, 0, stream>>>(b_ih1, pb1);

  dim3 pg(GDIM / 256, M / 64);  // (6, 512)
  const size_t rec_lds = (size_t)(192 + 32) * HIDDEN * sizeof(bf16);  // 224KB

  // Layer 0
  gemm_bias_bf16<<<pg, 256, 0, stream>>>(xbf, INPUT_PAD, wih0, INPUT_PAD, pb0, xgb, GDIM);
  zero_u32<<<2, 256, 0, stream>>>(bars, 600);
  gru_rec<<<REC_BLOCKS, 256, rec_lds, stream>>>(whh0, b_hh0, xgb, h_a, h_b, h32, ys0, bars);

  // Layer 1 (only final h needed)
  gemm_bias_bf16<<<pg, 256, 0, stream>>>(ys0, HIDDEN, wih1, HIDDEN, pb1, xgb, GDIM);
  zero_u32<<<2, 256, 0, stream>>>(bars, 600);
  gru_rec<<<REC_BLOCKS, 256, rec_lds, stream>>>(whh1, b_hh1, xgb, h_a, h_b, h32,
                                                (bf16*)nullptr, bars);

  // Head
  head_kernel<<<BATCH, 256, 0, stream>>>(h32, bn_w, bn_b, bn_mean, bn_var,
                                         fc1_w, fc1_b, ln_w, ln_b, fc2_w, fc2_b, out);
}
